// MatlabBlock_56264071577749
// MI455X (gfx1250) — matrix operand order repaired and checked by reading, not compiled, not run on hardware
//
#include <hip/hip_runtime.h>
#include <hip/hip_bf16.h>

// ---------------------------------------------------------------------------
// MatlabBlock cosine-attention, fused flash-style, bf16 WMMA on gfx1250.
//   N=4, C=256, B=4096
//   K = Wk X + Wk0 ; Q = Wq X + Wq0
//   Y[i,j] = <Q_i,K_j>/(|Q_i||K_j|) ; SM = softmax over i ; Z = X SM
// Fused as flash attention: queries = d, keys = i, values = X[:,i].
// ---------------------------------------------------------------------------

typedef __attribute__((ext_vector_type(16))) __bf16 v16bf;
typedef __attribute__((ext_vector_type(8)))  __bf16 v8bf;
typedef __attribute__((ext_vector_type(4)))  __bf16 v4bf;
typedef __attribute__((ext_vector_type(8)))  float  v8f;

#define NB 4
#define CC 256
#define BB 4096
#define TI 128                       // i-tile per flash step
#define SLD 130                      // Sld row pitch (floats)
#define PLD 136                      // PldT row pitch (bf16, mult of 8)

// ---- WMMA fragment loaders (wave32, 16x16x32 bf16) ------------------------
// A (16x32, MxK): lanes 0-15 -> M=lane, K = {0..7, 16..23};
//                 lanes 16-31 -> M=lane-16, K = {8..15, 24..31}
__device__ inline v16bf load_a16(const __bf16* A, int lda) {
  const int lane = threadIdx.x & 31;
  const int m    = lane & 15;
  const int kb   = (lane >> 4) << 3;          // 0 or 8
  const __bf16* p = A + (size_t)m * lda + kb;
  v8bf lo = *(const v8bf*)p;                  // K = kb .. kb+7
  v8bf hi = *(const v8bf*)(p + 16);           // K = kb+16 .. kb+23
  v16bf r;
#pragma unroll
  for (int e = 0; e < 8; ++e) { r[e] = lo[e]; r[8 + e] = hi[e]; }
  return r;
}

// B (32x16, KxN) sourced from a TRANSPOSED buffer: B[k][n] = Bt[n*ldb + k]
// lanes 0-15 -> N=lane, K = {0..7, 16..23}; lanes 16-31 -> N=lane-16, K = {8..15, 24..31}
// -> per-lane two 16-byte loads (works for global and LDS).
__device__ inline v16bf load_bt16(const __bf16* Bt, int ldb) {
  const int lane = threadIdx.x & 31;
  const int n    = lane & 15;
  const int kb   = (lane >> 4) << 3;          // 0 or 8
  const __bf16* p = Bt + (size_t)n * ldb + kb;
  v8bf lo = *(const v8bf*)p;
  v8bf hi = *(const v8bf*)(p + 16);
  v16bf r;
#pragma unroll
  for (int e = 0; e < 8; ++e) { r[e] = lo[e]; r[8 + e] = hi[e]; }
  return r;
}

__device__ inline v8f wmma_bf16(v16bf a, v16bf b, v8f c) {
  return __builtin_amdgcn_wmma_f32_16x16x32_bf16(false, a, false, b,
                                                 (short)0, c, false, false);
}

// ---- Kernel 0a: fp32 -> bf16 conversion (weights) -------------------------
__global__ void cvt_kernel(const float* __restrict__ src,
                           __bf16* __restrict__ dst, int count) {
  int i = (blockIdx.x * blockDim.x + threadIdx.x) * 4;
  if (i + 3 < count) {
    float4 v = *(const float4*)(src + i);
    dst[i + 0] = (__bf16)v.x; dst[i + 1] = (__bf16)v.y;
    dst[i + 2] = (__bf16)v.z; dst[i + 3] = (__bf16)v.w;
  }
}

// ---- Kernel 0b: X fp32 -> bf16, both [C][B] and transposed [B][C] ---------
__global__ __launch_bounds__(256)
void cvtx_kernel(const float* __restrict__ X, __bf16* __restrict__ Xb,
                 __bf16* __restrict__ XbT) {
  const int n  = blockIdx.z;
  const int c0 = blockIdx.y * 64;
  const int b0 = blockIdx.x * 64;
  __shared__ __bf16 T[64 * 68];
  const int tcol = threadIdx.x & 15;
  const int trow = threadIdx.x >> 4;

  const float* Xn  = X  + ((size_t)n * CC + c0) * BB + b0;
  __bf16*      Xbn = Xb + ((size_t)n * CC + c0) * BB + b0;
#pragma unroll
  for (int r = 0; r < 4; ++r) {
    const int c  = trow + r * 16;
    const int b4 = tcol * 4;
    float4 v = *(const float4*)(Xn + (size_t)c * BB + b4);
    v4bf w;
    w[0] = (__bf16)v.x; w[1] = (__bf16)v.y;
    w[2] = (__bf16)v.z; w[3] = (__bf16)v.w;
    *(v4bf*)(Xbn + (size_t)c * BB + b4) = w;
    *(v4bf*)(&T[c * 68 + b4]) = w;
  }
  __syncthreads();
  __bf16* XbTn = XbT + ((size_t)n * BB + b0) * CC + c0;
#pragma unroll
  for (int r = 0; r < 4; ++r) {
    const int b  = trow + r * 16;
    const int c4 = tcol * 4;
    v4bf w;
#pragma unroll
    for (int j = 0; j < 4; ++j) w[j] = T[(c4 + j) * 68 + b];
    *(v4bf*)(XbTn + (size_t)b * CC + c4) = w;
  }
}

// ---- Kernel 1: projections + per-column L2 normalization ------------------
// For a 64-wide column tile: P = W * X (+bias), n2[b] = sum_c P[c,b]^2,
// store P[c,b] * rsqrt(max(n2,eps)) transposed as [B][C] bf16.
__global__ __launch_bounds__(256)
void proj_kernel(const __bf16* __restrict__ Wkb, const __bf16* __restrict__ Wqb,
                 const __bf16* __restrict__ XbT,
                 const float* __restrict__ Wk0, const float* __restrict__ Wq0,
                 __bf16* __restrict__ KhT, __bf16* __restrict__ QhT) {
  const int n    = blockIdx.y;
  const int b0   = blockIdx.x * 64;
  const int wave = threadIdx.x >> 5;
  const int lane = threadIdx.x & 31;
  const int ncol = lane & 15;
  const int hi   = lane >> 4;
  const int c0   = wave * 32;                 // 8 waves cover c = 0..255

  __shared__ float norm2[64];

  const __bf16* XbTn = XbT + (size_t)n * BB * CC;

  for (int which = 0; which < 2; ++which) {
    const __bf16* W    = which ? Wqb : Wkb;
    const float*  bias = which ? Wq0 : Wk0;
    __bf16*       dstT = which ? QhT : KhT;

    v8f acc[2][4];
#pragma unroll
    for (int mt = 0; mt < 2; ++mt)
#pragma unroll
      for (int ct = 0; ct < 4; ++ct) acc[mt][ct] = (v8f){};

    for (int kk = 0; kk < 8; ++kk) {          // K loop over d = 0..255
      const int kc = kk * 32;
      v16bf bfr[4];
#pragma unroll
      for (int ct = 0; ct < 4; ++ct)          // B[k][n] = X[d=kc+k][b] = XbT[b][d]
        bfr[ct] = load_bt16(XbTn + (size_t)(b0 + ct * 16) * CC + kc, CC);
#pragma unroll
      for (int mt = 0; mt < 2; ++mt) {
        v16bf afr = load_a16(W + (size_t)(c0 + mt * 16) * CC + kc, CC);
#pragma unroll
        for (int ct = 0; ct < 4; ++ct)
          acc[mt][ct] = wmma_bf16(afr, bfr[ct], acc[mt][ct]);
      }
    }

    // bias (reference biases are zero-shaped [C,1] but honored anyway)
#pragma unroll
    for (int mt = 0; mt < 2; ++mt)
#pragma unroll
      for (int e = 0; e < 8; ++e) {
        const float bv = bias[c0 + mt * 16 + e + 8 * hi];
#pragma unroll
        for (int ct = 0; ct < 4; ++ct) acc[mt][ct][e] += bv;
      }

    if (threadIdx.x < 64) norm2[threadIdx.x] = 0.f;
    __syncthreads();
#pragma unroll
    for (int ct = 0; ct < 4; ++ct) {
      float s = 0.f;
#pragma unroll
      for (int mt = 0; mt < 2; ++mt)
#pragma unroll
        for (int e = 0; e < 8; ++e) s += acc[mt][ct][e] * acc[mt][ct][e];
      atomicAdd(&norm2[ct * 16 + ncol], s);
    }
    __syncthreads();

#pragma unroll
    for (int ct = 0; ct < 4; ++ct) {
      const float rn = rsqrtf(fmaxf(norm2[ct * 16 + ncol], 1e-12f));
      const int   b  = b0 + ct * 16 + ncol;
#pragma unroll
      for (int mt = 0; mt < 2; ++mt)
#pragma unroll
        for (int e = 0; e < 8; ++e) {
          const int c = c0 + mt * 16 + e + 8 * hi;
          dstT[((size_t)n * BB + b) * CC + c] = (__bf16)(acc[mt][ct][e] * rn);
        }
    }
    __syncthreads();                           // norm2 reused next pass
  }
}

// ---- Kernel 2: fused flash attention --------------------------------------
// Block: 64 d-columns; stream i in tiles of TI=128.
__global__ __launch_bounds__(256)
void attn_kernel(const __bf16* __restrict__ KhT, const __bf16* __restrict__ QhT,
                 const __bf16* __restrict__ Xb, float* __restrict__ Z) {
  const int n    = blockIdx.y;
  const int d0   = blockIdx.x * 64;
  const int wave = threadIdx.x >> 5;
  const int lane = threadIdx.x & 31;
  const int ncol = lane & 15;
  const int hi   = lane >> 4;

  __shared__ float  Sld[64 * SLD];             // S[d][i]
  __shared__ __bf16 PldT[64 * PLD];            // P transposed: PldT[d][i]
  __shared__ float  alphaArr[64];
  __shared__ float  lArr[64];

  const __bf16* KhTn = KhT + (size_t)n * BB * CC;
  const __bf16* QhTn = QhT + (size_t)n * BB * CC;
  const __bf16* Xn   = Xb  + (size_t)n * CC * BB;

  v8f o[2][4];                                 // O rows c=[wave*32, wave*32+32)
#pragma unroll
  for (int mt = 0; mt < 2; ++mt)
#pragma unroll
    for (int ct = 0; ct < 4; ++ct) o[mt][ct] = (v8f){};

  // online-softmax state, replicated in the 4 lanes that share row d
  const int srow = threadIdx.x >> 2;           // d row handled by this thread
  const int sq   = threadIdx.x & 3;            // quarter of the i range
  float mrun = -3.0e38f, lrun = 0.f;

  for (int it = 0; it < BB / TI; ++it) {
    const int i0 = it * TI;

    // prefetch next i-tile of Qh^T and X into cache while we compute
    if (it + 1 < BB / TI) {
      __builtin_prefetch(QhTn + (size_t)(i0 + TI) * CC + threadIdx.x * 128, 0, 0);
      __builtin_prefetch(Xn + (size_t)(wave * 32 + hi * 16 + ncol) * BB +
                             i0 + TI + (lane & 1) * 64, 0, 0);
    }

    // ---- score tile S[64d x 128i]: 32 WMMA tiles, 4 per wave ----
#pragma unroll
    for (int tt = 0; tt < 4; ++tt) {
      const int t  = wave * 4 + tt;
      const int di = t >> 3, ii = t & 7;
      v8f s = (v8f){};
      for (int kk = 0; kk < 8; ++kk) {         // K loop over c = 0..255
        const int kc = kk * 32;
        v16bf a = load_a16(KhTn + (size_t)(d0 + di * 16) * CC + kc, CC);
        v16bf b = load_bt16(QhTn + (size_t)(i0 + ii * 16) * CC + kc, CC);
        s = wmma_bf16(a, b, s);
      }
#pragma unroll
      for (int e = 0; e < 8; ++e)
        Sld[(di * 16 + e + 8 * hi) * SLD + ii * 16 + ncol] = s[e];
    }
    __syncthreads();

    // ---- online softmax over i for each d row (4 lanes per row) ----
    {
      const float* srp = &Sld[srow * SLD + sq * 32];
      float tmax = -3.0e38f;
#pragma unroll
      for (int i = 0; i < 32; ++i) tmax = fmaxf(tmax, srp[i]);
      tmax = fmaxf(tmax, __shfl_xor(tmax, 1));
      tmax = fmaxf(tmax, __shfl_xor(tmax, 2));
      const float mnew  = fmaxf(mrun, tmax);
      const float alpha = __expf(mrun - mnew);
      float ps = 0.f;
      __bf16* prp = &PldT[srow * PLD + sq * 32];
#pragma unroll
      for (int i = 0; i < 32; ++i) {
        const float p = __expf(srp[i] - mnew);
        ps += p;
        prp[i] = (__bf16)p;
      }
      ps += __shfl_xor(ps, 1);
      ps += __shfl_xor(ps, 2);
      lrun = lrun * alpha + ps;
      mrun = mnew;
      if (sq == 0) alphaArr[srow] = alpha;
    }
    __syncthreads();

    // ---- rescale running O, then O += X[:, i-tile] * P ----
    float al[4];
#pragma unroll
    for (int ct = 0; ct < 4; ++ct) al[ct] = alphaArr[ct * 16 + ncol];
#pragma unroll
    for (int mt = 0; mt < 2; ++mt)
#pragma unroll
      for (int ct = 0; ct < 4; ++ct)
#pragma unroll
        for (int e = 0; e < 8; ++e) o[mt][ct][e] *= al[ct];

#pragma unroll
    for (int kc2 = 0; kc2 < TI; kc2 += 32) {   // i-tile K loop
      v16bf bb[4];
#pragma unroll
      for (int ct = 0; ct < 4; ++ct)           // B[k][n] = P[i][d] = PldT[d][i]
        bb[ct] = load_bt16(&PldT[(ct * 16) * PLD + kc2], PLD);
#pragma unroll
      for (int mt = 0; mt < 2; ++mt) {
        v16bf a = load_a16(
            Xn + (size_t)(wave * 32 + mt * 16) * BB + i0 + kc2, BB);
#pragma unroll
        for (int ct = 0; ct < 4; ++ct)
          o[mt][ct] = wmma_bf16(a, bb[ct], o[mt][ct]);
      }
    }
    __syncthreads();                           // Sld/PldT reused next iter
  }

  if (sq == 0) lArr[srow] = lrun;
  __syncthreads();

#pragma unroll
  for (int ct = 0; ct < 4; ++ct) {
    const float rl = 1.f / lArr[ct * 16 + ncol];
    const int   d  = d0 + ct * 16 + ncol;
#pragma unroll
    for (int mt = 0; mt < 2; ++mt)
#pragma unroll
      for (int e = 0; e < 8; ++e) {
        const int c = wave * 32 + mt * 16 + e + 8 * hi;
        Z[((size_t)n * CC + c) * BB + d] = o[mt][ct][e] * rl;
      }
  }
}

// ---------------------------------------------------------------------------
extern "C" void kernel_launch(void* const* d_in, const int* in_sizes, int n_in,
                              void* d_out, int out_size, void* d_ws, size_t ws_size,
                              hipStream_t stream) {
  const float* X   = (const float*)d_in[0];
  const float* Wk  = (const float*)d_in[1];
  const float* Wq  = (const float*)d_in[2];
  const float* Wk0 = (const float*)d_in[3];
  const float* Wq0 = (const float*)d_in[4];
  float* Z = (float*)d_out;

  char* ws = (char*)d_ws;
  __bf16* Xb  = (__bf16*)ws; ws += (size_t)NB * CC * BB * sizeof(__bf16);
  __bf16* XbT = (__bf16*)ws; ws += (size_t)NB * BB * CC * sizeof(__bf16);
  __bf16* Wkb = (__bf16*)ws; ws += (size_t)CC * CC * sizeof(__bf16);
  __bf16* Wqb = (__bf16*)ws; ws += (size_t)CC * CC * sizeof(__bf16);
  __bf16* KhT = (__bf16*)ws; ws += (size_t)NB * BB * CC * sizeof(__bf16);
  __bf16* QhT = (__bf16*)ws; ws += (size_t)NB * BB * CC * sizeof(__bf16);

  dim3 gx(BB / 64, CC / 64, NB);
  cvtx_kernel<<<gx, 256, 0, stream>>>(X, Xb, XbT);
  cvt_kernel<<<(CC * CC) / (256 * 4), 256, 0, stream>>>(Wk, Wkb, CC * CC);
  cvt_kernel<<<(CC * CC) / (256 * 4), 256, 0, stream>>>(Wq, Wqb, CC * CC);

  dim3 g(BB / 64, NB);
  proj_kernel<<<g, 256, 0, stream>>>(Wkb, Wqb, XbT, Wk0, Wq0, KhT, QhT);
  attn_kernel<<<g, 256, 0, stream>>>(KhT, QhT, Xb, Z);
}
